// GNNDefectDetectionModel_23261542875616
// MI455X (gfx1250) — matrix operand order repaired and checked by reading, not compiled, not run on hardware
//
#include <hip/hip_runtime.h>

typedef __attribute__((ext_vector_type(16))) _Float16 v16h;
typedef __attribute__((ext_vector_type(8)))  _Float16 v8h;
typedef __attribute__((ext_vector_type(8)))  float    v8f;

#define CDIM 128

// ---------------- utility kernels ----------------

__global__ void k_fill_zero(float* p, int n) {
  int i = blockIdx.x * blockDim.x + threadIdx.x;
  if (i < n) p[i] = 0.0f;
}

__global__ void k_degree(const int* __restrict__ dst, float* __restrict__ deg, int E) {
  int i = blockIdx.x * blockDim.x + threadIdx.x;
  if (i < E) unsafeAtomicAdd(&deg[dst[i]], 1.0f);
}

__global__ void k_finish_dis(float* d, int n) {
  int i = blockIdx.x * blockDim.x + threadIdx.x;
  if (i < n) d[i] = rsqrtf(d[i] + 1.0f);
}

__global__ void k_convert_f16(const float* __restrict__ in, _Float16* __restrict__ out, int n) {
  int i = blockIdx.x * blockDim.x + threadIdx.x;
  if (i < n) out[i] = (_Float16)in[i];
}

// W is [K, Ncols] row-major f32; WT is [Ncols, K] row-major f16
__global__ void k_transpose_convert(const float* __restrict__ W, _Float16* __restrict__ WT,
                                    int K, int Ncols) {
  int i = blockIdx.x * blockDim.x + threadIdx.x;
  if (i < K * Ncols) {
    int k = i / Ncols, n = i % Ncols;
    WT[n * K + k] = (_Float16)W[i];
  }
}

// ---------------- WMMA GEMM: C[M, NT*16] = A[M,128] x B[128, NT*16] ----------------
// A: f16 row-major [M,128]; BT: f16 [NT*16, 128] (B transposed, row-major).
// B is staged once per workgroup into LDS (padded stride -> conflict-free ds_load_b128),
// then each wave computes a 16-row x NT*16-col strip with f32-accumulating WMMAs.
template <int NT>
__global__ void k_gemm_wmma(const _Float16* __restrict__ A, const _Float16* __restrict__ BT,
                            float* __restrict__ Cout, int M,
                            const float* __restrict__ bias, int relu) {
  const int K   = 128;
  const int LDB = K + 8;                       // padded stride (halves): +16B per row
  __shared__ _Float16 Bl[NT * 16 * LDB];

  // cooperative copy of BT (NT*16 rows x 128 halves) into padded LDS
  const int chunks = NT * 16 * 16;             // 16-byte (8-half) chunks
  for (int ch = threadIdx.x; ch < chunks; ch += blockDim.x) {
    int col = ch >> 4, ci = ch & 15;
    *(v8h*)(Bl + col * LDB + ci * 8) = *(const v8h*)(BT + (size_t)col * K + ci * 8);
  }
  __syncthreads();

  int lane = threadIdx.x & 31;
  int wave = threadIdx.x >> 5;
  int rowTile = blockIdx.x * 8 + wave;
  if (rowTile * 16 >= M) return;               // after barrier; wave-uniform

  int r16  = lane & 15;
  int hsel = lane >> 4;

  v8f zero = {0.f, 0.f, 0.f, 0.f, 0.f, 0.f, 0.f, 0.f};
  v8f acc[NT];
#pragma unroll
  for (int n = 0; n < NT; ++n) acc[n] = zero;

  const _Float16* arow = A + (size_t)(rowTile * 16 + r16) * K;

#pragma unroll
  for (int kt = 0; kt < 4; ++kt) {
    // A tile 16x32: lane<16 holds K = [kb..kb+7] and [kb+16..kb+23]; lane>=16 offset +8
    int kbA = kt * 32 + hsel * 8;
    v8h alo = *(const v8h*)(arow + kbA);
    v8h ahi = *(const v8h*)(arow + kbA + 16);
    v16h a;
#pragma unroll
    for (int j = 0; j < 8; ++j) { a[j] = alo[j]; a[j + 8] = ahi[j]; }

    int kbB = kt * 32 + hsel * 8;              // B: K = [kb..kb+7] and [kb+16..kb+23]; lanes>=16 -> +8
#pragma unroll
    for (int n = 0; n < NT; ++n) {
      const _Float16* bcol = Bl + (n * 16 + r16) * LDB + kbB;
      v8h blo = *(const v8h*)(bcol);
      v8h bhi = *(const v8h*)(bcol + 16);
      v16h b;
#pragma unroll
      for (int j = 0; j < 8; ++j) { b[j] = blo[j]; b[j + 8] = bhi[j]; }

      acc[n] = __builtin_amdgcn_wmma_f32_16x16x32_f16(
          false, a, false, b, (short)0, acc[n], false, false);
    }
  }

  // C/D layout: VGPR rr -> row rr (lanes 0-15) / rr+8 (lanes 16-31), col = lane&15
  int rowBase = rowTile * 16 + hsel * 8;
  int ld = NT * 16;
#pragma unroll
  for (int n = 0; n < NT; ++n) {
    int col = n * 16 + r16;
    float bv = bias ? bias[col] : 0.0f;
#pragma unroll
    for (int rr = 0; rr < 8; ++rr) {
      float v = acc[n][rr] + bv;
      if (relu) v = fmaxf(v, 0.0f);
      Cout[(size_t)(rowBase + rr) * ld + col] = v;
    }
  }
}

// ---------------- GCN aggregation ----------------

// Ag[i,c] = dis[i]^2 * Hf[i,c] + bias[c]
__global__ void k_self_init(const float* __restrict__ Hf, const float* __restrict__ dis,
                            const float* __restrict__ bias, float* __restrict__ Ag, int total) {
  int i = blockIdx.x * blockDim.x + threadIdx.x;
  if (i < total) {
    int row = i >> 7;
    float d = dis[row];
    Ag[i] = d * d * Hf[i] + bias[i & (CDIM - 1)];
  }
}

// Ag[dst] += dis[src]*dis[dst] * Hf[src]  (one thread per edge x 4 columns)
__global__ void k_edge_scatter(const int* __restrict__ src, const int* __restrict__ dst,
                               const float* __restrict__ dis, const float* __restrict__ Hf,
                               float* __restrict__ Ag, int E) {
  int t = blockIdx.x * blockDim.x + threadIdx.x;
  if (t >= E * 32) return;
  int e = t >> 5;
  int c = (t & 31) * 4;
  int s = src[e], d = dst[e];
  float w = dis[s] * dis[d];
  const float4 v = *(const float4*)(Hf + (size_t)s * CDIM + c);
  float* p = Ag + (size_t)d * CDIM + c;
  unsafeAtomicAdd(p + 0, v.x * w);
  unsafeAtomicAdd(p + 1, v.y * w);
  unsafeAtomicAdd(p + 2, v.z * w);
  unsafeAtomicAdd(p + 3, v.w * w);
}

// ---------------- BatchNorm ----------------

// one block per column: compute scale = g*rsqrt(var+eps), shift = be - mu*scale
__global__ void k_bn_stats(const float* __restrict__ Ag, const float* __restrict__ gam,
                           const float* __restrict__ bet, float* __restrict__ scale,
                           float* __restrict__ shift, int n) {
  int c = blockIdx.x;
  float s = 0.f, s2 = 0.f;
  for (int r = threadIdx.x; r < n; r += blockDim.x) {
    float v = Ag[(size_t)r * CDIM + c];
    s += v; s2 += v * v;
  }
  __shared__ float sh[256], sh2[256];
  sh[threadIdx.x] = s; sh2[threadIdx.x] = s2;
  __syncthreads();
  for (int st = 128; st > 0; st >>= 1) {
    if (threadIdx.x < st) { sh[threadIdx.x] += sh[threadIdx.x + st];
                            sh2[threadIdx.x] += sh2[threadIdx.x + st]; }
    __syncthreads();
  }
  if (threadIdx.x == 0) {
    float mu  = sh[0] / n;
    float var = sh2[0] / n - mu * mu;
    float rs  = rsqrtf(var + 1e-5f);
    float sc  = gam[c] * rs;
    scale[c] = sc;
    shift[c] = bet[c] - mu * sc;
  }
}

// y = relu(x*scale + shift); write f32 (for pool) and f16 (next layer A operand)
__global__ void k_bn_apply_relu(const float* __restrict__ Ag, const float* __restrict__ scale,
                                const float* __restrict__ shift, float* __restrict__ Hf,
                                _Float16* __restrict__ Xh, int total) {
  int i = blockIdx.x * blockDim.x + threadIdx.x;
  if (i < total) {
    int c = i & (CDIM - 1);
    float v = fmaxf(Ag[i] * scale[c] + shift[c], 0.0f);
    Hf[i] = v;
    Xh[i] = (_Float16)v;
  }
}

// ---------------- pooling + head ----------------

__global__ void k_pool(const float* __restrict__ Hf, const int* __restrict__ batch,
                       float* __restrict__ psum, float* __restrict__ pcnt, int N) {
  int t = blockIdx.x * blockDim.x + threadIdx.x;
  if (t >= N * 32) return;
  int i = t >> 5;
  int q = t & 31;
  int c = q * 4;
  int g = batch[i];
  const float4 v = *(const float4*)(Hf + (size_t)i * CDIM + c);
  float* p = psum + (size_t)g * CDIM + c;
  unsafeAtomicAdd(p + 0, v.x);
  unsafeAtomicAdd(p + 1, v.y);
  unsafeAtomicAdd(p + 2, v.z);
  unsafeAtomicAdd(p + 3, v.w);
  if (q == 0) unsafeAtomicAdd(&pcnt[g], 1.0f);
}

__global__ void k_pool_div(float* __restrict__ psum, const float* __restrict__ pcnt, int total) {
  int i = blockIdx.x * blockDim.x + threadIdx.x;
  if (i < total) psum[i] /= fmaxf(pcnt[i >> 7], 1.0f);
}

__global__ void k_fc3(const float* __restrict__ z2, const float* __restrict__ w,
                      const float* __restrict__ b, float* __restrict__ out, int G) {
  int g = blockIdx.x * blockDim.x + threadIdx.x;
  if (g < G) {
    float s = b[0];
#pragma unroll
    for (int k = 0; k < 64; ++k) s += z2[(size_t)g * 64 + k] * w[k];
    out[g] = s;
  }
}

// ---------------- launch ----------------

extern "C" void kernel_launch(void* const* d_in, const int* in_sizes, int n_in,
                              void* d_out, int out_size, void* d_ws, size_t ws_size,
                              hipStream_t stream) {
  const float* x      = (const float*)d_in[0];
  const int*   eidx   = (const int*)d_in[1];
  const int*   batch  = (const int*)d_in[2];
  const float* Wl[3]  = {(const float*)d_in[4], (const float*)d_in[6], (const float*)d_in[8]};
  const float* bl[3]  = {(const float*)d_in[5], (const float*)d_in[7], (const float*)d_in[9]};
  const float* gl[3]  = {(const float*)d_in[10], (const float*)d_in[12], (const float*)d_in[14]};
  const float* bel[3] = {(const float*)d_in[11], (const float*)d_in[13], (const float*)d_in[15]};
  const float* fc1w = (const float*)d_in[16];
  const float* fc1b = (const float*)d_in[17];
  const float* fc2w = (const float*)d_in[18];
  const float* fc2b = (const float*)d_in[19];
  const float* fc3w = (const float*)d_in[20];
  const float* fc3b = (const float*)d_in[21];

  const int N = in_sizes[2];
  const int E = in_sizes[1] / 2;
  const int G = 256;
  const int* src = eidx;
  const int* dst = eidx + E;

  // workspace bump allocator (256B aligned)
  char*  base = (char*)d_ws;
  size_t off  = 0;
  auto alloc = [&](size_t bytes) -> char* {
    off = (off + 255) & ~(size_t)255;
    char* p = base + off;
    off += bytes;
    return p;
  };
  float*     dis   = (float*)alloc((size_t)N * 4);
  float*     Hf    = (float*)alloc((size_t)N * CDIM * 4);
  float*     Ag    = (float*)alloc((size_t)N * CDIM * 4);
  _Float16*  Xh    = (_Float16*)alloc((size_t)N * CDIM * 2);
  _Float16*  WhT   = (_Float16*)alloc((size_t)CDIM * CDIM * 2);
  float*     scale = (float*)alloc(CDIM * 4);
  float*     shift = (float*)alloc(CDIM * 4);
  float*     psum  = (float*)alloc((size_t)G * CDIM * 4);
  float*     pcnt  = (float*)alloc((size_t)G * 4);
  _Float16*  Ph    = (_Float16*)alloc((size_t)G * CDIM * 2);
  float*     z1    = (float*)alloc((size_t)G * CDIM * 4);
  _Float16*  z1h   = (_Float16*)alloc((size_t)G * CDIM * 2);
  float*     z2    = (float*)alloc((size_t)G * 64 * 4);

  const int NC  = N * CDIM;
  const int TB  = 256;
  auto blocks = [&](long long n) { return (int)((n + TB - 1) / TB); };

  // degree -> dis = rsqrt(deg+1)
  k_fill_zero<<<blocks(N), TB, 0, stream>>>(dis, N);
  k_degree<<<blocks(E), TB, 0, stream>>>(dst, dis, E);
  k_finish_dis<<<blocks(N), TB, 0, stream>>>(dis, N);

  // x -> f16
  k_convert_f16<<<blocks(NC), TB, 0, stream>>>(x, Xh, NC);

  const int gemmGridN = (N / 16 + 7) / 8;   // N multiple of 16
  const int gemmGridG = (G / 16 + 7) / 8;

  for (int l = 0; l < 3; ++l) {
    k_transpose_convert<<<blocks(CDIM * CDIM), TB, 0, stream>>>(Wl[l], WhT, CDIM, CDIM);
    k_gemm_wmma<8><<<gemmGridN, TB, 0, stream>>>(Xh, WhT, Hf, N, nullptr, 0);
    k_self_init<<<blocks(NC), TB, 0, stream>>>(Hf, dis, bl[l], Ag, NC);
    k_edge_scatter<<<blocks((long long)E * 32), TB, 0, stream>>>(src, dst, dis, Hf, Ag, E);
    k_bn_stats<<<CDIM, TB, 0, stream>>>(Ag, gl[l], bel[l], scale, shift, N);
    k_bn_apply_relu<<<blocks(NC), TB, 0, stream>>>(Ag, scale, shift, Hf, Xh, NC);
  }

  // global mean pool
  k_fill_zero<<<blocks(G * CDIM), TB, 0, stream>>>(psum, G * CDIM);
  k_fill_zero<<<blocks(G), TB, 0, stream>>>(pcnt, G);
  k_pool<<<blocks((long long)N * 32), TB, 0, stream>>>(Hf, batch, psum, pcnt, N);
  k_pool_div<<<blocks(G * CDIM), TB, 0, stream>>>(psum, pcnt, G * CDIM);
  k_convert_f16<<<blocks(G * CDIM), TB, 0, stream>>>(psum, Ph, G * CDIM);

  // fc1: [G,128] x [128,128] + b, relu
  k_transpose_convert<<<blocks(CDIM * CDIM), TB, 0, stream>>>(fc1w, WhT, CDIM, CDIM);
  k_gemm_wmma<8><<<gemmGridG, TB, 0, stream>>>(Ph, WhT, z1, G, fc1b, 1);
  k_convert_f16<<<blocks(G * CDIM), TB, 0, stream>>>(z1, z1h, G * CDIM);

  // fc2: [G,128] x [128,64] + b, relu
  k_transpose_convert<<<blocks(CDIM * 64), TB, 0, stream>>>(fc2w, WhT, CDIM, 64);
  k_gemm_wmma<4><<<gemmGridG, TB, 0, stream>>>(z1h, WhT, z2, G, fc2b, 1);

  // fc3: [G,64] x [64,1] + b
  k_fc3<<<1, TB, 0, stream>>>(z2, fc3w, fc3b, (float*)d_out, G);

  (void)n_in; (void)out_size; (void)ws_size;
}
